// MQA_22436909154690
// MI455X (gfx1250) — compile-verified, matrix operand order checked by reading, not run on hardware
//
#include <hip/hip_runtime.h>
#include <cstdint>
#include <cstddef>

// ---------------------------------------------------------------------------
// MQA reference collapses: softmax over a singleton axis == 1.0, so
//   out[b,l,h*128+d] = v[b,l,d],  v = x @ Wv^T + bv,  Wv = Wkv[128:256,:]
// Only live work: GEMM (M=16384, N=128, K=2048) + 16x head broadcast.
// Strategy: bf16 WMMA (f32 accum), LDS double buffering, async global->LDS
// copies for the weight tiles (CDNA5 path), coalesced b128 output stores.
// ---------------------------------------------------------------------------

#ifndef __has_builtin
#define __has_builtin(x) 0
#endif

#if __has_builtin(__builtin_amdgcn_global_load_async_to_lds_b128)
#define USE_ASYNC_LDS_COPY 1
#else
#define USE_ASYNC_LDS_COPY 0
#endif

typedef __attribute__((ext_vector_type(16))) __bf16 v16bf;
typedef __attribute__((ext_vector_type(8)))  __bf16 v8bf;
typedef __attribute__((ext_vector_type(4)))  __bf16 v4bf;
typedef __attribute__((ext_vector_type(8)))  float  v8f;
typedef __attribute__((ext_vector_type(4)))  float  f32x4;
typedef __attribute__((ext_vector_type(4)))  int    v4i;

constexpr int kB  = 4, kL = 4096, kE = 2048, kH = 16, kHD = 128;
constexpr int kM  = kB * kL;       // 16384 rows
constexpr int kK  = kE;            // 2048 reduction
constexpr int MT  = 64;            // M tile per workgroup
constexpr int KS  = 64;            // K step per iteration
constexpr int NWAVE = 8;           // 256 threads = 8 wave32
constexpr int KSTEPS = kK / KS;    // 32

__device__ __forceinline__ v4bf cvt4(f32x4 f) {
  v4bf r;
  r[0] = (__bf16)f[0]; r[1] = (__bf16)f[1];
  r[2] = (__bf16)f[2]; r[3] = (__bf16)f[3];
  return r;
}

// 16-byte global -> LDS copy (async on CDNA5; sync fallback otherwise)
__device__ __forceinline__ void copyB128(const __bf16* g, __bf16* l) {
#if USE_ASYNC_LDS_COPY
  __builtin_amdgcn_global_load_async_to_lds_b128(
      (__attribute__((address_space(1))) v4i*)g,
      (__attribute__((address_space(3))) v4i*)l, 0, 0);
#else
  *(v8bf*)l = *(const v8bf*)g;
#endif
}

__device__ __forceinline__ void waitAsync4() {
#if USE_ASYNC_LDS_COPY
#if __has_builtin(__builtin_amdgcn_s_wait_asynccnt)
  __builtin_amdgcn_s_wait_asynccnt(4);
#else
  asm volatile("s_wait_asynccnt 4" ::: "memory");
#endif
#endif
}

__device__ __forceinline__ void waitAsync0() {
#if USE_ASYNC_LDS_COPY
#if __has_builtin(__builtin_amdgcn_s_wait_asynccnt)
  __builtin_amdgcn_s_wait_asynccnt(0);
#else
  asm volatile("s_wait_asynccnt 0" ::: "memory");
#endif
#endif
}

// Load a 16-element bf16 fragment: two contiguous 8-elem (16B) runs at p, p+16.
// Matches the ISA 16-bit A/B striping (lane>>4 selects the K-half, handled by
// the caller via the +hi8*8 base offset).
__device__ __forceinline__ v16bf ld_frag(const __bf16* p) {
  v8bf lo = *(const v8bf*)p;
  v8bf hi = *(const v8bf*)(p + 16);
  v16bf r;
#pragma unroll
  for (int i = 0; i < 8; ++i) { r[i] = lo[i]; r[i + 8] = hi[i]; }
  return r;
}

// ---- kernel 0: convert Wv = Wkv[128:256, :] (fp32) to bf16 in workspace ----
__global__ void __launch_bounds__(256)
wv_bf16_kernel(const float* __restrict__ Wkv, __bf16* __restrict__ Wb) {
  int i = blockIdx.x * 256 + threadIdx.x;            // over 128*2048/4 vec4s
  f32x4 w = ((const f32x4*)(Wkv + (size_t)kHD * kE))[i];
  ((v4bf*)Wb)[i] = cvt4(w);
}

// ---- kernel 1: V = x @ Wv^T + bv via bf16 WMMA, broadcast across heads ----
__global__ void __launch_bounds__(256)
mqa_v_gemm_kernel(const float* __restrict__ x,
                  const __bf16* __restrict__ Wb,   // [128][2048] bf16
                  const float* __restrict__ bkv,   // bias; v part at +128
                  float* __restrict__ out) {
  // LDS carving: A 2 bufs * 64*64 bf16 = 16 KB; B 2 bufs * 8 waves * 16*64
  // bf16 = 32 KB. Reused as the 64x128 fp32 V tile (32 KB) in the epilogue.
  __shared__ alignas(16) unsigned char lds_raw[49152];
  __bf16* const Abuf = (__bf16*)lds_raw;                 // [2][MT*KS]
  __bf16* const Bbuf = (__bf16*)(lds_raw + 16384);       // [2][NWAVE][16*KS]

  const int tid  = threadIdx.x;
  const int wave = tid >> 5;
  const int lane = tid & 31;
  const int nl   = lane & 15;
  const int hi8  = lane >> 4;
  const size_t m0 = (size_t)blockIdx.x * MT;

  // x-tile loader mapping: thread -> (row 0..63, 16-float column segment)
  const int arow = tid >> 2;
  const int acol = (tid & 3) * 16;
  const float* const agbase = x + (m0 + arow) * (size_t)kK + acol;

  f32x4 areg[4];
#pragma unroll
  for (int j = 0; j < 4; ++j) areg[j] = ((const f32x4*)agbase)[j];

  auto issueB = [&](int buf, int kg) {
    __bf16* lbase = Bbuf + buf * (NWAVE * 16 * KS) + wave * (16 * KS);
#pragma unroll
    for (int j = 0; j < 4; ++j) {
      int linear = j * 32 + lane;          // 16 rows * 8 16B-segments
      int r   = linear >> 3;
      int seg = linear & 7;
      copyB128(Wb + (size_t)(wave * 16 + r) * kK + kg + seg * 8,
               lbase + r * KS + seg * 8);
    }
  };
  auto storeA = [&](int buf) {
    v4bf* d = (v4bf*)(Abuf + buf * (MT * KS) + arow * KS + acol);
#pragma unroll
    for (int j = 0; j < 4; ++j) d[j] = cvt4(areg[j]);
  };

  issueB(0, 0);
  storeA(0);

  v8f acc[4] = {};   // 4 M-subtiles of 16x16 f32

  for (int i = 0; i < KSTEPS; ++i) {
    const bool have_next = (i + 1) < KSTEPS;
    if (have_next) {
      const int kg = (i + 1) * KS;
      const float* ag = agbase + kg;
#pragma unroll
      for (int j = 0; j < 4; ++j) areg[j] = ((const f32x4*)ag)[j];
      issueB((i + 1) & 1, kg);
      waitAsync4();                 // oldest 4 (this step's B tile) done
    } else {
      waitAsync0();
    }
    __syncthreads();                // A tile of this step visible to all waves

    const __bf16* ab = Abuf + (i & 1) * (MT * KS);
    const __bf16* bb = Bbuf + (i & 1) * (NWAVE * 16 * KS) + wave * (16 * KS);
#pragma unroll
    for (int kb = 0; kb < 2; ++kb) {
      v16bf bfrag = ld_frag(bb + nl * KS + kb * 32 + hi8 * 8);
#pragma unroll
      for (int ms = 0; ms < 4; ++ms) {
        v16bf afrag = ld_frag(ab + (ms * 16 + nl) * KS + kb * 32 + hi8 * 8);
        acc[ms] = __builtin_amdgcn_wmma_f32_16x16x32_bf16(
            false, afrag, false, bfrag, (short)0, acc[ms], false, false);
      }
    }
    if (have_next) storeA((i + 1) & 1);
  }

  __syncthreads();   // all LDS reads done; reuse lds_raw as fp32 V tile

  const float bn = bkv[kHD + wave * 16 + nl];
  float* const Vt = (float*)lds_raw;           // [MT][kHD]
#pragma unroll
  for (int ms = 0; ms < 4; ++ms) {
#pragma unroll
    for (int r = 0; r < 8; ++r) {
      // D layout: VGPR r holds M=r (lanes 0-15) / M=8+r (lanes 16-31)
      int m = ms * 16 + hi8 * 8 + r;
      Vt[m * kHD + wave * 16 + nl] = acc[ms][r] + bn;
    }
  }
  __syncthreads();

  // out[m, h*128 + d] = V[m, d] for all 16 heads; coalesced b128 stores.
  for (int it = 0; it < 128; ++it) {
    int linear = it * 256 + tid;              // 64 rows * 512 float4/row
    int m   = linear >> 9;
    int seg = linear & 511;                   // float4 index within out row
    f32x4 v = *(const f32x4*)(Vt + m * kHD + (seg & 31) * 4);
    *(f32x4*)(out + (m0 + m) * (size_t)kE + (size_t)seg * 4) = v;
  }
}

extern "C" void kernel_launch(void* const* d_in, const int* in_sizes, int n_in,
                              void* d_out, int out_size, void* d_ws, size_t ws_size,
                              hipStream_t stream) {
  // setup_inputs order: x(0), Wq(1), bq(2), Wkv(3), bkv(4) — Wq/bq are dead.
  const float* x   = (const float*)d_in[0];
  const float* Wkv = (const float*)d_in[3];
  const float* bkv = (const float*)d_in[4];
  float* out = (float*)d_out;
  __bf16* Wb = (__bf16*)d_ws;                 // 128*2048*2 = 512 KB scratch

  wv_bf16_kernel<<<(kHD * kK / 4) / 256, 256, 0, stream>>>(Wkv, Wb);
  mqa_v_gemm_kernel<<<kM / MT, 256, 0, stream>>>(x, Wb, bkv, out);

  (void)in_sizes; (void)n_in; (void)out_size; (void)ws_size;
}
